// MPNN_20684562498181
// MI455X (gfx1250) — matrix operand order repaired and checked by reading, not compiled, not run on hardware
//
#include <hip/hip_runtime.h>
#include <hip/hip_bf16.h>

// ---------------------------------------------------------------------------
// MPNN on MI455X (gfx1250): 48 sequential message-passing steps, each a fused
// f16 WMMA GEMM  h_new = relu(h@U1 + h[idx]@(Vw@U2) + e@(Ew@U3) + bias),
// then a WMMA readout GEMM + deterministic reduction + tiny MLP head.
// ---------------------------------------------------------------------------

#define N_NODES 200000
#define DEG     16
#define T_ROUNDS 3
#define HSTR    80            // padded f16 node-state row stride (halves)
#define KTOT    192           // padded K for step GEMM (6 ksteps of 32)
#define KRO     96            // padded K for readout GEMM (3 ksteps of 32)
#define NTILES_STEP 5         // 80 output cols
#define NTILES_RO   8         // 128 output cols
#define TILES   (N_NODES / 16)        // 12500
#define WAVES_PER_BLOCK 8
#define STEP_BLOCKS ((TILES + WAVES_PER_BLOCK - 1) / WAVES_PER_BLOCK)  // 1563

typedef __attribute__((ext_vector_type(16))) _Float16 v16h;
typedef __attribute__((ext_vector_type(8)))  float    v8f;
typedef __attribute__((ext_vector_type(4)))  unsigned int vec4u;

union Frag {
    v16h  h;
    vec4u q[2];
};

// ---------------------------------------------------------------------------
// Precompute fused weights:
//   WcT[n][k] (f16, [80][192]) : k<70 -> Uw[k][n]; k in [80,150) -> (Vw@U2)[k-80][n];
//                                k in [160,166) -> (Ew@U3)[k-160][n]; else 0
//   biasc[n]  (f32, [80])      : Ub + Vb@U2 + Eb@U3 (0 for n>=70)
//   RcT[c][k] (f16, [128][96]) : k<70 -> Rw[k][c] + Rw[k+70][c]; else 0
// ---------------------------------------------------------------------------
__global__ __launch_bounds__(256) void mpnn_prep_kernel(
    const float* __restrict__ Vw, const float* __restrict__ Vb,
    const float* __restrict__ Ew, const float* __restrict__ Eb,
    const float* __restrict__ Uw, const float* __restrict__ Ub,
    const float* __restrict__ Rw, const float* __restrict__ Rb,
    _Float16* __restrict__ WcT, float* __restrict__ biasc,
    _Float16* __restrict__ RcT)
{
    const int tid = threadIdx.x;

    for (int i = tid; i < 80 * KTOT; i += 256) {
        const int n = i / KTOT;
        const int k = i % KTOT;
        float val = 0.f;
        if (n < 70) {
            if (k < 70) {
                val = Uw[k * 70 + n];
            } else if (k >= 80 && k < 150) {
                const int kk = k - 80;
                float s = 0.f;
                for (int q = 0; q < 70; ++q)
                    s += Vw[kk * 70 + q] * Uw[(70 + q) * 70 + n];
                val = s;
            } else if (k >= 160 && k < 166) {
                const int kk = k - 160;
                float s = 0.f;
                for (int q = 0; q < 6; ++q)
                    s += Ew[kk * 6 + q] * Uw[(140 + q) * 70 + n];
                val = s;
            }
        }
        WcT[n * KTOT + k] = (_Float16)val;
    }

    for (int n = tid; n < 80; n += 256) {
        float val = 0.f;
        if (n < 70) {
            val = Ub[n];
            for (int q = 0; q < 70; ++q) val += Vb[q] * Uw[(70 + q) * 70 + n];
            for (int q = 0; q < 6;  ++q) val += Eb[q] * Uw[(140 + q) * 70 + n];
        }
        biasc[n] = val;
    }

    for (int i = tid; i < 128 * KRO; i += 256) {
        const int c = i / KRO;
        const int k = i % KRO;
        float val = (k < 70) ? (Rw[k * 128 + c] + Rw[(k + 70) * 128 + c]) : 0.f;
        RcT[c * KRO + k] = (_Float16)val;
    }
}

// ---------------------------------------------------------------------------
// Pad + convert initial state: f32 [N][70] -> f16 [N][80] (cols 70..79 = 0)
// ---------------------------------------------------------------------------
__global__ __launch_bounds__(256) void mpnn_padh_kernel(
    const float* __restrict__ h, _Float16* __restrict__ hout)
{
    const long i = (long)blockIdx.x * blockDim.x + threadIdx.x;
    if (i >= (long)N_NODES * HSTR) return;
    const long v = i / HSTR;
    const int  k = (int)(i % HSTR);
    hout[i] = (k < 70) ? (_Float16)h[v * 70 + k] : (_Float16)0.f;
}

// ---------------------------------------------------------------------------
// One message-passing step (neighbor slot j), all nodes.
// Each wave: one 16-node tile; A fragments from global f16 state (aligned
// 16B chunks), B fragments from LDS-resident fused weights, 30 WMMAs.
// ---------------------------------------------------------------------------
__global__ __launch_bounds__(256) void mpnn_step_kernel(
    const _Float16* __restrict__ hin, _Float16* __restrict__ hout,
    const int* __restrict__ nbr, const float* __restrict__ ef,
    const _Float16* __restrict__ WcT, const float* __restrict__ biasc,
    int j)
{
    __shared__ __align__(16) _Float16 sW[80 * KTOT];   // 30720 B
    {
        const vec4u* src = (const vec4u*)WcT;
        vec4u* dst = (vec4u*)sW;
        for (int i = threadIdx.x; i < (80 * KTOT) / 8; i += 256) dst[i] = src[i];
    }
    __syncthreads();

    const int lane = threadIdx.x & 31;
    const int wave = threadIdx.x >> 5;
    const int tile = blockIdx.x * WAVES_PER_BLOCK + wave;
    if (tile >= TILES) return;                 // wave-uniform

    const int row  = lane & 15;
    const int half = lane >> 4;
    const int v    = tile * 16 + row;
    const int nb   = nbr[v * DEG + j];
    const _Float16* hv = hin + (size_t)v  * HSTR;
    const _Float16* hg = hin + (size_t)nb * HSTR;

    const vec4u zq = {0u, 0u, 0u, 0u};
    Frag A[6];
    #pragma unroll
    for (int k = 0; k < 5; ++k) {
        const int c0 = k * 32 + half * 8;
        const int c1 = c0 + 16;
        const _Float16* p0 = (c0 < 80) ? (hv + c0) : (hg + (c0 - 80));
        const _Float16* p1 = (c1 < 80) ? (hv + c1) : (hg + (c1 - 80));
        A[k].q[0] = *(const vec4u*)p0;
        A[k].q[1] = *(const vec4u*)p1;
    }
    // kstep 5: K[160..166) = edge features, rest zero
    A[5].q[0] = zq;
    A[5].q[1] = zq;
    if (half == 0) {
        const float* ep = ef + ((size_t)v * DEG + j) * 6;
        alignas(16) _Float16 t[8];
        t[0] = (_Float16)ep[0]; t[1] = (_Float16)ep[1]; t[2] = (_Float16)ep[2];
        t[3] = (_Float16)ep[3]; t[4] = (_Float16)ep[4]; t[5] = (_Float16)ep[5];
        t[6] = (_Float16)0.f;   t[7] = (_Float16)0.f;
        A[5].q[0] = *(const vec4u*)t;
    }

    const v8f zc = {0.f, 0.f, 0.f, 0.f, 0.f, 0.f, 0.f, 0.f};
    v8f C[NTILES_STEP];
    #pragma unroll
    for (int t = 0; t < NTILES_STEP; ++t) C[t] = zc;

    #pragma unroll
    for (int k = 0; k < 6; ++k) {
        #pragma unroll
        for (int t = 0; t < NTILES_STEP; ++t) {
            Frag B;
            const vec4u* bp =
                (const vec4u*)(sW + (t * 16 + row) * KTOT + k * 32 + half * 8);
            B.q[0] = bp[0];
            B.q[1] = bp[2];
            C[t] = __builtin_amdgcn_wmma_f32_16x16x32_f16(
                false, A[k].h, false, B.h, (short)0, C[t], false, false);
        }
    }

    // Epilogue: bias + relu + f16 store (C layout: col = t*16+row, row = p+8*half)
    #pragma unroll
    for (int t = 0; t < NTILES_STEP; ++t) {
        const int n = t * 16 + row;
        const float bn = biasc[n];
        #pragma unroll
        for (int p = 0; p < 8; ++p) {
            const int M = p + half * 8;
            float val = C[t][p] + bn;
            val = val > 0.f ? val : 0.f;
            hout[(size_t)(tile * 16 + M) * HSTR + n] = (_Float16)val;
        }
    }
}

// ---------------------------------------------------------------------------
// Readout: per-block partial of fm[c] = sum_v relu(h[v] @ Rc + Rb)[c]
// WMMA over K=96, 8 column tiles; deterministic (no atomics).
// ---------------------------------------------------------------------------
__global__ __launch_bounds__(256) void mpnn_readout_kernel(
    const _Float16* __restrict__ hin, const _Float16* __restrict__ RcT,
    const float* __restrict__ Rb, float* __restrict__ partials)
{
    __shared__ __align__(16) _Float16 sR[128 * KRO];   // 24576 B
    __shared__ float sFm[WAVES_PER_BLOCK * 128];       //  4096 B
    {
        const vec4u* src = (const vec4u*)RcT;
        vec4u* dst = (vec4u*)sR;
        for (int i = threadIdx.x; i < (128 * KRO) / 8; i += 256) dst[i] = src[i];
        for (int i = threadIdx.x; i < WAVES_PER_BLOCK * 128; i += 256) sFm[i] = 0.f;
    }
    __syncthreads();

    const int lane = threadIdx.x & 31;
    const int wave = threadIdx.x >> 5;
    const int tile = blockIdx.x * WAVES_PER_BLOCK + wave;
    const int row  = lane & 15;
    const int half = lane >> 4;

    if (tile < TILES) {                         // wave-uniform
        const _Float16* hv = hin + (size_t)(tile * 16 + row) * HSTR;
        const vec4u zq = {0u, 0u, 0u, 0u};
        Frag A[3];
        #pragma unroll
        for (int k = 0; k < 3; ++k) {
            const int c0 = k * 32 + half * 8;
            const int c1 = c0 + 16;
            A[k].q[0] = (c0 < HSTR) ? *(const vec4u*)(hv + c0) : zq;
            A[k].q[1] = (c1 < HSTR) ? *(const vec4u*)(hv + c1) : zq;
        }

        const v8f zc = {0.f, 0.f, 0.f, 0.f, 0.f, 0.f, 0.f, 0.f};
        v8f C[NTILES_RO];
        #pragma unroll
        for (int t = 0; t < NTILES_RO; ++t) C[t] = zc;

        #pragma unroll
        for (int k = 0; k < 3; ++k) {
            #pragma unroll
            for (int t = 0; t < NTILES_RO; ++t) {
                Frag B;
                const vec4u* bp =
                    (const vec4u*)(sR + (t * 16 + row) * KRO + k * 32 + half * 8);
                B.q[0] = bp[0];
                B.q[1] = bp[2];
                C[t] = __builtin_amdgcn_wmma_f32_16x16x32_f16(
                    false, A[k].h, false, B.h, (short)0, C[t], false, false);
            }
        }

        #pragma unroll
        for (int t = 0; t < NTILES_RO; ++t) {
            const int n = t * 16 + row;
            const float bn = Rb[n];
            float s = 0.f;
            #pragma unroll
            for (int p = 0; p < 8; ++p) {
                float val = C[t][p] + bn;
                s += (val > 0.f ? val : 0.f);
            }
            // combine rows 0..7 (half 0) with rows 8..15 (half 1) of same column
            s += __shfl_xor(s, 16, 32);
            if (half == 0) sFm[wave * 128 + n] = s;
        }
    }
    __syncthreads();

    if (threadIdx.x < 128) {
        float s = 0.f;
        #pragma unroll
        for (int w = 0; w < WAVES_PER_BLOCK; ++w) s += sFm[w * 128 + threadIdx.x];
        partials[(size_t)blockIdx.x * 128 + threadIdx.x] = s;
    }
}

// ---------------------------------------------------------------------------
// Head: fixed-order reduction of block partials, then the tiny MLP.
// ---------------------------------------------------------------------------
__global__ __launch_bounds__(128) void mpnn_head_kernel(
    const float* __restrict__ partials, int nblocks,
    const float* __restrict__ S1w, const float* __restrict__ S1b,
    const float* __restrict__ S2w, const float* __restrict__ S2b,
    const float* __restrict__ Hw,  const float* __restrict__ Hb,
    const float* __restrict__ Ow,  const float* __restrict__ Ob,
    float* __restrict__ out)
{
    __shared__ float fm[128], a[128], b[100], c2[100];
    const int tid = threadIdx.x;

    {
        float s = 0.f;
        for (int bix = 0; bix < nblocks; ++bix)
            s += partials[(size_t)bix * 128 + tid];
        fm[tid] = s;
    }
    __syncthreads();

    {
        float s = S1b[tid];
        for (int k = 0; k < 128; ++k) s += fm[k] * S1w[k * 128 + tid];
        a[tid] = s > 0.f ? s : 0.f;
    }
    __syncthreads();

    if (tid < 100) {
        float s = S2b[tid];
        for (int k = 0; k < 128; ++k) s += a[k] * S2w[k * 100 + tid];
        b[tid] = s;
    }
    __syncthreads();

    if (tid < 100) {
        float s = Hb[tid];
        for (int k = 0; k < 100; ++k) s += b[k] * Hw[k * 100 + tid];
        c2[tid] = s > 0.f ? s : 0.f;
    }
    __syncthreads();

    if (tid == 0) {
        float s = Ob[0];
        for (int k = 0; k < 100; ++k) s += c2[k] * Ow[k];
        out[0] = s;
    }
}

// ---------------------------------------------------------------------------
extern "C" void kernel_launch(void* const* d_in, const int* in_sizes, int n_in,
                              void* d_out, int out_size, void* d_ws, size_t ws_size,
                              hipStream_t stream) {
    (void)in_sizes; (void)n_in; (void)out_size; (void)ws_size;

    const float* h    = (const float*)d_in[0];
    const float* ef   = (const float*)d_in[1];
    const int*   nbr  = (const int*)  d_in[2];
    const float* Vw   = (const float*)d_in[3];
    const float* Vb   = (const float*)d_in[4];
    const float* Ew   = (const float*)d_in[5];
    const float* Eb   = (const float*)d_in[6];
    const float* Uw   = (const float*)d_in[7];
    const float* Ub   = (const float*)d_in[8];
    const float* Rw   = (const float*)d_in[9];
    const float* Rb   = (const float*)d_in[10];
    const float* S1w  = (const float*)d_in[11];
    const float* S1b  = (const float*)d_in[12];
    const float* S2w  = (const float*)d_in[13];
    const float* S2b  = (const float*)d_in[14];
    const float* Hw   = (const float*)d_in[15];
    const float* Hb   = (const float*)d_in[16];
    const float* Ow   = (const float*)d_in[17];
    const float* Ob   = (const float*)d_in[18];
    float* out = (float*)d_out;

    // Workspace layout (all 16B-aligned offsets)
    char* ws = (char*)d_ws;
    const size_t szH = (size_t)N_NODES * HSTR * sizeof(_Float16);   // 32,000,000
    _Float16* hA   = (_Float16*)(ws);
    _Float16* hB   = (_Float16*)(ws + szH);
    _Float16* WcT  = (_Float16*)(ws + 2 * szH);                      // 30,720 B
    _Float16* RcT  = (_Float16*)(ws + 2 * szH + 30720);              // 24,576 B
    float*    biasc= (float*)   (ws + 2 * szH + 30720 + 24576);      //    320 B
    float*    parts= (float*)   (ws + 2 * szH + 30720 + 24576 + 320);// 1563*128*4

    mpnn_prep_kernel<<<1, 256, 0, stream>>>(Vw, Vb, Ew, Eb, Uw, Ub, Rw, Rb,
                                            WcT, biasc, RcT);
    mpnn_padh_kernel<<<(N_NODES * HSTR) / 256, 256, 0, stream>>>(h, hA);

    _Float16* cur = hA;
    _Float16* nxt = hB;
    for (int s = 0; s < T_ROUNDS * DEG; ++s) {
        mpnn_step_kernel<<<STEP_BLOCKS, 256, 0, stream>>>(
            cur, nxt, nbr, ef, WcT, biasc, s & (DEG - 1));
        _Float16* tmp = cur; cur = nxt; nxt = tmp;
    }

    mpnn_readout_kernel<<<STEP_BLOCKS, 256, 0, stream>>>(cur, RcT, Rb, parts);
    mpnn_head_kernel<<<1, 128, 0, stream>>>(parts, STEP_BLOCKS,
                                            S1w, S1b, S2w, S2b, Hw, Hb, Ow, Ob, out);
}
